// Smodel_41721312313477
// MI455X (gfx1250) — matrix operand order repaired and checked by reading, compiled, not run on hardware
//
#include <hip/hip_runtime.h>
#include <hip/hip_bf16.h>

// ---------------------------------------------------------------------------
// Types for CDNA5 WMMA (gfx1250, wave32): D(16x16 f32) = A(16x32 bf16) x B(32x16 bf16) + C
// ---------------------------------------------------------------------------
typedef __attribute__((ext_vector_type(16))) __bf16 v16bf;
typedef __attribute__((ext_vector_type(8)))  float  v8f;

union FragU { uint4 q[2]; v16bf v; };

__device__ __forceinline__ unsigned short f2bf(float f) {
    unsigned int u = __float_as_uint(f);
    unsigned int r = u + 0x7FFFu + ((u >> 16) & 1u);   // round-to-nearest-even
    return (unsigned short)(r >> 16);
}

// ---------------------------------------------------------------------------
// K0: zero output, zero stats, fill perm with -1
// ---------------------------------------------------------------------------
__global__ void k_init(float* __restrict__ out, long outN,
                       float* __restrict__ stats, int statsN,
                       int* __restrict__ perm, int permN) {
    long gid = (long)blockIdx.x * 256 + threadIdx.x;
    if (gid < outN)   out[gid]   = 0.f;
    if (gid < statsN) stats[gid] = 0.f;
    if (gid < permN)  perm[gid]  = -1;
}

// ---------------------------------------------------------------------------
// K1: per-triplet geometry features -> bf16 [E,64] (56 used + 8 zero pad),
//     branch flag, per-block branch0 counts
// ---------------------------------------------------------------------------
__global__ void k_geo_prep(const float* __restrict__ coords,
                           const float* __restrict__ face_x,
                           const int* __restrict__ eidx,
                           const int* __restrict__ edx_ij,
                           const int* __restrict__ edx_jk,
                           const int* __restrict__ ewf,
                           unsigned short* __restrict__ geo_raw,
                           unsigned char* __restrict__ flags,
                           int* __restrict__ blkcnt, int E) {
    __shared__ int cnt;
    const int tid = threadIdx.x;
    const int e = blockIdx.x * 256 + tid;
    if (tid == 0) cnt = 0;
    __syncthreads();
    int isb0 = 0;
    if (e < E) {
        const int fij = ewf[edx_ij[e]];
        const int fjk = ewf[edx_jk[e]];
        const int vi = eidx[e];
        const int vj = eidx[(size_t)E + e];
        const int vk = eidx[(size_t)2 * E + e];
        const float cjx = coords[(size_t)vj*3+0], cjy = coords[(size_t)vj*3+1], cjz = coords[(size_t)vj*3+2];
        const float cix = coords[(size_t)vi*3+0], ciy = coords[(size_t)vi*3+1], ciz = coords[(size_t)vi*3+2];
        const float ckx = coords[(size_t)vk*3+0], cky = coords[(size_t)vk*3+1], ckz = coords[(size_t)vk*3+2];
        const float v1x = cjx-cix, v1y = cjy-ciy, v1z = cjz-ciz;   // cj - ci
        const float v2x = ckx-cjx, v2y = cky-cjy, v2z = ckz-cjz;   // ck - cj
        const float dij = sqrtf(v1x*v1x + v1y*v1y + v1z*v1z);
        const float djk = sqrtf(v2x*v2x + v2y*v2y + v2z*v2z);      // ||cj-ck|| == ||ck-cj||
        const float crx = v1y*v2z - v1z*v2y;
        const float cry = v1z*v2x - v1x*v2z;
        const float crz = v1x*v2y - v1y*v2x;
        const float cn  = sqrtf(crx*crx + cry*cry + crz*crz);
        const float dp  = v1x*v2x + v1y*v2y + v1z*v2z;
        const float theta = atan2f(cn, dp);
        unsigned short* dst = geo_raw + (size_t)e * 64;
        dst[0] = f2bf(dij); dst[1] = f2bf(djk); dst[2] = f2bf(theta); dst[3] = 0;
        dst[4] = f2bf(cjx); dst[5] = f2bf(cjy); dst[6] = f2bf(cjz);
        dst[7] = f2bf(cix); dst[8] = f2bf(ciy); dst[9] = f2bf(ciz);
        const float* fa = face_x + (size_t)fij * 23;
        const float* fb = face_x + (size_t)fjk * 23;
        #pragma unroll
        for (int q = 0; q < 23; ++q) dst[10 + q] = f2bf(fa[q]);
        #pragma unroll
        for (int q = 0; q < 23; ++q) dst[33 + q] = f2bf(fb[q]);
        #pragma unroll
        for (int q = 0; q < 8; ++q) dst[56 + q] = 0;
        const int brv = (fij == fjk) ? 0 : 1;
        flags[e] = (unsigned char)brv;
        isb0 = 1 - brv;
    }
    if (isb0) atomicAdd(&cnt, 1);
    __syncthreads();
    if (tid == 0) blkcnt[blockIdx.x] = cnt;
}

// ---------------------------------------------------------------------------
// K2: sequential scan of per-block counts -> partition offsets + header
//     hdr[0]=c0  hdr[1]=c0 padded to 256  hdr[2]=end of branch1 region
// ---------------------------------------------------------------------------
__global__ void k_scan(const int* __restrict__ blkcnt, int* __restrict__ o0,
                       int* __restrict__ o1, int* __restrict__ hdr, int NB, int E) {
    if (threadIdx.x != 0 || blockIdx.x != 0) return;
    int off0 = 0;
    for (int b = 0; b < NB; ++b) { o0[b] = off0; off0 += blkcnt[b]; }
    const int c0 = off0;
    const int c0p = (c0 + 255) & ~255;
    int off1 = c0p;
    for (int b = 0; b < NB; ++b) {
        int rows = E - b * 256; if (rows > 256) rows = 256;
        o1[b] = off1;
        off1 += rows - blkcnt[b];
    }
    hdr[0] = c0; hdr[1] = c0p; hdr[2] = off1;
}

// ---------------------------------------------------------------------------
// K3: stable scatter of row ids into the branch-partitioned permutation
// ---------------------------------------------------------------------------
__global__ void k_scatter(const unsigned char* __restrict__ flags,
                          const int* __restrict__ o0, const int* __restrict__ o1,
                          int* __restrict__ perm, int E) {
    __shared__ int fl[256];
    __shared__ int pre[256];
    const int tid = threadIdx.x;
    const int e = blockIdx.x * 256 + tid;
    int flag = 2;
    if (e < E) flag = flags[e];
    fl[tid] = (flag == 0) ? 1 : 0;
    __syncthreads();
    if (tid == 0) {
        int s = 0;
        for (int q = 0; q < 256; ++q) { pre[q] = s; s += fl[q]; }
    }
    __syncthreads();
    if (e < E) {
        int dest = (flag == 0) ? (o0[blockIdx.x] + pre[tid])
                               : (o1[blockIdx.x] + (tid - pre[tid]));
        perm[dest] = e;
    }
}

// ---------------------------------------------------------------------------
// Unified WMMA GEMM layer kernel.  256 threads = 8 waves, 128 rows/block.
//   mode 0: input = raw bf16 [row,64] natural order (geo layer), unmasked stats
//   mode 1: input = gathered concat [nf[i] nf[j] nf[k] geo_act] via perm (Kin=256)
//   mode 2: input = f32 pre-activation + fused BN(scale,shift)+ReLU  (Kin=64)
// Emits y = X*W^T + b to yout (f32) and masked per-feature sum / sumsq stats.
// ---------------------------------------------------------------------------
__global__ __launch_bounds__(256) void k_gemm(
    int mode, int nChunks, int Kw,
    const unsigned short* __restrict__ xb,
    const float* __restrict__ xf,
    const int* __restrict__ perm,
    const int* __restrict__ eidx,
    const float* __restrict__ Wg,
    const float* __restrict__ biasg,
    int wStride, int bStride,
    const float* __restrict__ statsIn,
    float* __restrict__ statsOut,
    const int* __restrict__ hdr,
    float* __restrict__ yout,
    int E, int useBranch) {
    __shared__ __align__(16) unsigned short Xl[8][16][72];   // 72-stride: bank-conflict free
    __shared__ __align__(16) unsigned short Wl[64][72];
    __shared__ float aS[64], cS[64];
    __shared__ float redS[128];

    const int tid  = threadIdx.x;
    const int wave = tid >> 5;
    const int lane = tid & 31;
    const int m = lane & 15;       // row (A) / column (B,D) within tile
    const int h = lane >> 4;       // half-wave
    const int rowBase = blockIdx.x * 128;

    const int c0  = hdr[0];
    const int c0p = hdr[1];
    const int tot = hdr[2];
    const int branch = (useBranch && rowBase >= c0p) ? 1 : 0;

    const float* W    = Wg + (size_t)branch * wStride;
    const float* bias = biasg + (size_t)branch * bStride;
    float* sOut = statsOut + branch * 256;

    if (tid < 128) redS[tid] = 0.f;
    if (mode == 2 && tid < 64) {
        const float* sIn = statsIn + branch * 256;
        aS[tid] = sIn[128 + tid];
        cS[tid] = sIn[192 + tid];
    }

    v8f acc[4];
    #pragma unroll
    for (int nt = 0; nt < 4; ++nt)
        #pragma unroll
        for (int r = 0; r < 8; ++r) acc[nt][r] = 0.f;

    for (int chunk = 0; chunk < nChunks; ++chunk) {
        __syncthreads();
        // ---- stage weight slice W[n][chunk*64 .. +64) as bf16 B-matrix source
        for (int s = tid; s < 4096; s += 256) {
            const int n = s >> 6, kk = s & 63;
            const int kg = chunk * 64 + kk;
            const float w = (kg < Kw) ? W[(size_t)n * Kw + kg] : 0.f;
            Wl[n][kk] = f2bf(w);
        }
        // ---- stage 16 rows x 64 cols of activations per wave (2 lanes/row)
        {
            const int r0 = lane & 15;
            const int hf = lane >> 4;
            const int p  = rowBase + wave * 16 + r0;
            unsigned short* dst = &Xl[wave][r0][32 * hf];
            if (mode == 0) {
                if (p < E) {
                    const uint4* s4 = (const uint4*)(xb + (size_t)p * 64 + 32 * hf);
                    uint4* d4 = (uint4*)dst;
                    #pragma unroll
                    for (int q = 0; q < 4; ++q) d4[q] = s4[q];
                } else {
                    #pragma unroll
                    for (int q = 0; q < 32; ++q) dst[q] = 0;
                }
            } else if (mode == 1) {
                const int e = perm[p];
                if (chunk < 3) {                       // nf[i], nf[j], nf[k]
                    if (e >= 0) {
                        const int node = eidx[(size_t)chunk * E + e];
                        const float* src = xf + (size_t)node * 64 + 32 * hf;
                        #pragma unroll
                        for (int q = 0; q < 32; ++q) dst[q] = f2bf(src[q]);
                    } else {
                        #pragma unroll
                        for (int q = 0; q < 32; ++q) dst[q] = 0;
                    }
                } else {                               // geo_act (already bf16)
                    if (e >= 0) {
                        const uint4* s4 = (const uint4*)(xb + (size_t)e * 64 + 32 * hf);
                        uint4* d4 = (uint4*)dst;
                        #pragma unroll
                        for (int q = 0; q < 4; ++q) d4[q] = s4[q];
                    } else {
                        #pragma unroll
                        for (int q = 0; q < 32; ++q) dst[q] = 0;
                    }
                }
            } else {                                   // fused BN-apply + ReLU
                const float* src = xf + (size_t)p * 64 + 32 * hf;
                #pragma unroll
                for (int q = 0; q < 32; ++q) {
                    const int col = 32 * hf + q;
                    float v = fmaf(src[q], aS[col], cS[col]);
                    dst[q] = f2bf(fmaxf(v, 0.f));
                }
            }
        }
        __syncthreads();
        // ---- two K=32 WMMA steps over this 64-deep chunk
        #pragma unroll
        for (int ks = 0; ks < 2; ++ks) {
            FragU a;   // A 16x32 bf16: lane m row, K = 8h+e (e<8) / 16+8h+(e-8)
            a.q[0] = *(const uint4*)&Xl[wave][m][ks * 32 + 8 * h];
            a.q[1] = *(const uint4*)&Xl[wave][m][ks * 32 + 16 + 8 * h];
            #pragma unroll
            for (int nt = 0; nt < 4; ++nt) {
                FragU b;  // B 32x16 bf16: lane col, K = 8h+e (e<8) / 16+8h+(e-8)
                const int col = nt * 16 + m;
                b.q[0] = *(const uint4*)&Wl[col][ks * 32 + 8 * h];
                b.q[1] = *(const uint4*)&Wl[col][ks * 32 + 16 + 8 * h];
                acc[nt] = __builtin_amdgcn_wmma_f32_16x16x32_bf16(
                    false, a.v, false, b.v, (short)0, acc[nt], false, false);
            }
        }
    }
    // ---- epilogue: bias, y store (f32), masked per-feature stats
    #pragma unroll
    for (int nt = 0; nt < 4; ++nt) {
        const int f = nt * 16 + m;
        const float bv = bias[f];
        float s1 = 0.f, s2 = 0.f;
        #pragma unroll
        for (int r = 0; r < 8; ++r) {
            const int row = rowBase + wave * 16 + 8 * h + r;   // D: M = r + 8*half
            const float y = acc[nt][r] + bv;
            yout[(size_t)row * 64 + f] = y;
            const bool valid = useBranch ? ((row < c0) || (row >= c0p && row < tot))
                                         : (row < E);
            if (valid) { s1 += y; s2 += y * y; }
        }
        atomicAdd(&redS[f], s1);
        atomicAdd(&redS[64 + f], s2);
    }
    __syncthreads();
    if (tid < 128) atomicAdd(&sOut[tid], redS[tid]);
}

// ---------------------------------------------------------------------------
// Finalize BN stats: fold (gamma, beta, mean, var) into scale a / shift c
// ---------------------------------------------------------------------------
__global__ void k_finalize(float* __restrict__ slot,
                           const float* __restrict__ g, const float* __restrict__ b,
                           int gStride, const int* __restrict__ hdr,
                           int E, int nSlots, int useBranch) {
    const int tid = threadIdx.x;
    if (tid >= 64 * nSlots) return;
    const int bi = tid >> 6, f = tid & 63;
    float* s = slot + bi * 256;
    float cnt;
    if (!useBranch) cnt = (float)E;
    else { const int c0 = hdr[0]; cnt = (float)(bi ? (E - c0) : c0); }
    cnt = fmaxf(cnt, 1.f);
    const float mean = s[f] / cnt;
    const float var  = fmaxf(s[64 + f] / cnt - mean * mean, 0.f);
    const float a = g[bi * gStride + f] * rsqrtf(var + 1e-5f);
    const float c = b[bi * gStride + f] - mean * a;
    s[128 + f] = a;
    s[192 + f] = c;
}

// ---------------------------------------------------------------------------
// Apply geo BN+ReLU once, cache as bf16 (reused by all 3 interaction rounds)
// ---------------------------------------------------------------------------
__global__ void k_geo_apply(const float* __restrict__ y, const float* __restrict__ slot,
                            unsigned short* __restrict__ gact, long n) {
    const long gid = (long)blockIdx.x * 256 + threadIdx.x;
    if (gid >= n) return;
    const int f = (int)(gid & 63);
    const float v = y[gid] * slot[128 + f] + slot[192 + f];
    gact[gid] = f2bf(fmaxf(v, 0.f));
}

// ---------------------------------------------------------------------------
// Combine: BN-apply + ReLU (leaky-ReLU after ReLU == identity) * att[branch],
// then atomic segment-sum onto node i  (output slice doubles as next nf)
// ---------------------------------------------------------------------------
__global__ void k_combine(const float* __restrict__ y, const float* __restrict__ slot,
                          const int* __restrict__ perm, const int* __restrict__ idx_i,
                          const float* __restrict__ att, const int* __restrict__ hdr,
                          float* __restrict__ out, int RP, int E) {
    const int gid = blockIdx.x * 256 + threadIdx.x;
    const int p = gid >> 2;
    if (p >= RP) return;
    const int c0 = hdr[0], c0p = hdr[1], tot = hdr[2];
    const bool b1 = p >= c0p;
    if (!b1 && p >= c0) return;
    if (b1 && p >= tot) return;
    const int e = perm[p];
    if (e < 0) return;
    const int node = idx_i[e];
    const int fb = (gid & 3) * 16;
    const float* s = slot + (b1 ? 256 : 0);
    const float am = att[b1 ? 1 : 0];
    const float* yr = y + (size_t)p * 64 + fb;
    float* o = out + (size_t)node * 64 + fb;
    #pragma unroll
    for (int q = 0; q < 16; ++q) {
        float v = yr[q] * s[128 + fb + q] + s[192 + fb + q];
        v = fmaxf(v, 0.f);
        atomicAdd(&o[q], v * am);
    }
}

// ---------------------------------------------------------------------------
// Host orchestration
// ---------------------------------------------------------------------------
extern "C" void kernel_launch(void* const* d_in, const int* in_sizes, int n_in,
                              void* d_out, int out_size, void* d_ws, size_t ws_size,
                              hipStream_t stream) {
    (void)out_size;
    if (n_in < 19) return;
    const float* input_feature = (const float*)d_in[0];
    const float* coords = (const float*)d_in[1];
    const float* face_x = (const float*)d_in[3];   // face_norm unused (angle_face * 0)
    const float* att  = (const float*)d_in[4];
    const float* W_geo = (const float*)d_in[5];
    const float* b_geo = (const float*)d_in[6];
    const float* g_geo = (const float*)d_in[7];
    const float* be_geo = (const float*)d_in[8];
    const float* W0 = (const float*)d_in[9];
    const float* b0 = (const float*)d_in[10];
    const float* Wr = (const float*)d_in[11];
    const float* br = (const float*)d_in[12];
    const float* bn_g = (const float*)d_in[13];
    const float* bn_b = (const float*)d_in[14];
    const int* ewf = (const int*)d_in[15];
    const int* eidx = (const int*)d_in[16];
    const int* edx_jk = (const int*)d_in[17];
    const int* edx_ij = (const int*)d_in[18];

    const int H = 64;
    const int N = in_sizes[0] / H;
    const int E = in_sizes[17];
    const int NB  = (E + 255) / 256;
    const int nbG = (E + 127) / 128;
    const int nbP = (E + 256 + 127) / 128;     // covers padded partition (<= E+256 rows)
    const int RP  = nbP * 128;

    // ---- workspace carving (ybufA overlays geo_raw; disjoint lifetimes)
    char* ws = (char*)d_ws;
    size_t cur = 0;
    auto carve = [&](size_t bytes) {
        size_t o = cur; cur = (cur + bytes + 255) & ~(size_t)255; return o;
    };
    const size_t off_hdr   = carve(64 * 4);
    const size_t off_stats = carve((size_t)26 * 256 * 4);
    const size_t off_blk   = carve((size_t)3 * NB * 4);
    const size_t off_flags = carve((size_t)E);
    const size_t off_perm  = carve((size_t)RP * 4);
    const size_t off_gact  = carve((size_t)E * 64 * 2);
    const size_t szA = (size_t)RP * 64 * 4;
    const size_t szG = (size_t)E * 64 * 2;
    const size_t off_A = carve(szA > szG ? szA : szG);
    const size_t off_B = carve((size_t)RP * 64 * 4);
    if (cur > ws_size) return;   // workspace too small — bail safely

    int* hdr = (int*)(ws + off_hdr);
    float* stats = (float*)(ws + off_stats);
    int* blkcnt = (int*)(ws + off_blk);
    int* o0 = blkcnt + NB;
    int* o1 = blkcnt + 2 * NB;
    unsigned char* flags = (unsigned char*)(ws + off_flags);
    int* perm = (int*)(ws + off_perm);
    unsigned short* geo_act = (unsigned short*)(ws + off_gact);
    unsigned short* geo_raw = (unsigned short*)(ws + off_A);
    float* ybufA = (float*)(ws + off_A);
    float* ybufB = (float*)(ws + off_B);
    float* out = (float*)d_out;

    const long outN = 3L * N * H;
    long initN = outN;
    if (RP > initN) initN = RP;
    if (26 * 256 > initN) initN = 26 * 256;

    k_init<<<(unsigned)((initN + 255) / 256), 256, 0, stream>>>(out, outN, stats, 26 * 256, perm, RP);
    k_geo_prep<<<NB, 256, 0, stream>>>(coords, face_x, eidx, edx_ij, edx_jk, ewf,
                                       geo_raw, flags, blkcnt, E);
    k_scan<<<1, 1, 0, stream>>>(blkcnt, o0, o1, hdr, NB, E);
    k_scatter<<<NB, 256, 0, stream>>>(flags, o0, o1, perm, E);

    // geo layer: [E,64(pad)] x W_geo[64,56]^T, unmasked BN stats (slot 0)
    k_gemm<<<nbG, 256, 0, stream>>>(0, 1, 56, geo_raw, nullptr, nullptr, nullptr,
                                    W_geo, b_geo, 0, 0, nullptr, stats, hdr, ybufB, E, 0);
    k_finalize<<<1, 128, 0, stream>>>(stats, g_geo, be_geo, 0, hdr, E, 1, 0);
    k_geo_apply<<<(unsigned)(((long)E * 64 + 255) / 256), 256, 0, stream>>>(
        ybufB, stats, geo_act, (long)E * 64);

    for (int t = 0; t < 3; ++t) {
        const float* nf = (t == 0) ? input_feature : (out + (size_t)(t - 1) * N * H);
        const int slotL0 = 1 + (t * 4 + 0) * 2;
        // layer 0: gathered [nf_i | nf_j | nf_k | geo] (K=256) -> ybufA
        k_gemm<<<nbP, 256, 0, stream>>>(1, 4, 256, geo_act, nf, perm, eidx,
            W0 + (size_t)t * 2 * 64 * 256, b0 + (size_t)t * 2 * 64, 64 * 256, 64,
            nullptr, stats + slotL0 * 256, hdr, ybufA, E, 1);
        k_finalize<<<1, 128, 0, stream>>>(stats + slotL0 * 256,
            bn_g + (size_t)(t * 2 * 4 + 0) * 64, bn_b + (size_t)(t * 2 * 4 + 0) * 64,
            4 * 64, hdr, E, 2, 1);
        // hidden layers 1..3 (Wr l=0..2), ping-pong A<->B
        for (int l = 0; l < 3; ++l) {
            const int slotIn = 1 + (t * 4 + l) * 2;
            const int slotOut = slotIn + 2;
            float* yin = (l % 2 == 0) ? ybufA : ybufB;
            float* yo  = (l % 2 == 0) ? ybufB : ybufA;
            k_gemm<<<nbP, 256, 0, stream>>>(2, 1, 64, nullptr, yin, nullptr, nullptr,
                Wr + (size_t)((t * 2) * 3 + l) * 64 * 64,
                br + (size_t)((t * 2) * 3 + l) * 64, 3 * 64 * 64, 3 * 64,
                stats + slotIn * 256, stats + slotOut * 256, hdr, yo, E, 1);
            k_finalize<<<1, 128, 0, stream>>>(stats + slotOut * 256,
                bn_g + (size_t)(t * 2 * 4 + l + 1) * 64,
                bn_b + (size_t)(t * 2 * 4 + l + 1) * 64, 4 * 64, hdr, E, 2, 1);
        }
        // BN-apply + ReLU + att, segment-sum into out[t] (also next round's nf)
        const int slotFinal = 1 + (t * 4 + 3) * 2;
        k_combine<<<(unsigned)(((long)RP * 4 + 255) / 256), 256, 0, stream>>>(
            ybufB, stats + slotFinal * 256, perm, eidx, att, hdr,
            out + (size_t)t * N * H, RP, E);
    }
}
